// ImprovedTransformerEncoderBlock_52364241272888
// MI455X (gfx1250) — matrix operand order repaired and checked by reading, not compiled, not run on hardware
//
#include <hip/hip_runtime.h>
#include <hip/hip_bf16.h>

// ---------------------------------------------------------------------------
// ImprovedTransformerEncoderBlock for MI455X (gfx1250, wave32, WMMA).
// All heavy contractions go through v_wmma_f32_16x16x32_bf16 (f32 accum).
// GEMM operand panels are staged into LDS with GLOBAL_LOAD_ASYNC_TO_LDS_B128
// (ASYNCcnt path), double-buffered against the WMMA consume loop.
// ---------------------------------------------------------------------------

typedef __attribute__((ext_vector_type(8)))  float  v8f;
typedef __attribute__((ext_vector_type(16))) __bf16 v16bf;

#define D_MODEL 1024
#define N_HEADS 16
#define D_K     64
#define D_FF    4096
#define MAX_REL 32
#define NREL    (2 * MAX_REL + 1)   // 65
#define BATCH   4
#define SEQ     1024
#define M_ROWS  (BATCH * SEQ)       // 4096

union FragBF {
    v16bf  v;
    uint4  q[2];
    __bf16 h[16];
};

__device__ __forceinline__ v8f wmma_bf16(v16bf a, v16bf b, v8f c) {
    // D = A(16x32) * B(32x16) + C, f32 accumulate
    return __builtin_amdgcn_wmma_f32_16x16x32_bf16(
        /*neg_a=*/false, a, /*neg_b=*/false, b,
        /*c_mod=*/(short)0, c, /*reuse_a=*/false, /*reuse_b=*/false);
}

// Async 16B copy global -> LDS (tracked by ASYNCcnt, per CDNA5 ISA).
__device__ __forceinline__ void async_ld16(const __bf16* g, __bf16* l) {
    unsigned lds_addr = (unsigned)(unsigned long long)l;
    asm volatile("global_load_async_to_lds_b128 %0, %1, off"
                 :: "v"(lds_addr), "v"((unsigned long long)g)
                 : "memory");
}

template <int N>
__device__ __forceinline__ void wait_async() {
    asm volatile("s_wait_asynccnt %0" :: "n"(N) : "memory");
}

// A-matrix 16x32 bf16 fragment (row-major source, leading dim lda).
// Lane l: M row = (l&15); elements 0..7 at K = k0 + 8*(l>=16) + e,
// elements 8..15 at K = k0 + 16 + 8*(l>=16) + (e-8). Two 16B loads.
__device__ __forceinline__ v16bf load_a_frag(const __bf16* __restrict__ base,
                                             int lda, int m0, int k0, int lane) {
    const __bf16* p = base + (size_t)(m0 + (lane & 15)) * lda + k0 + ((lane >> 4) << 3);
    FragBF f;
    f.q[0] = *(const uint4*)p;
    f.q[1] = *(const uint4*)(p + 16);
    return f.v;
}

// B-matrix 32x16 bf16 fragment where B = W^T and W is (N x K) row-major.
// Lane l: N = n0 + (l&15); elements e=0..15 at K = k0 + 16*(l>=16) + e.
__device__ __forceinline__ v16bf load_b_frag(const __bf16* __restrict__ base,
                                             int ldb, int n0, int k0, int lane) {
    const __bf16* p = base + (size_t)(n0 + (lane & 15)) * ldb + k0 + ((lane >> 4) << 4);
    FragBF f;
    f.q[0] = *(const uint4*)p;
    f.q[1] = *(const uint4*)(p + 8);
    return f.v;
}

// Same fragment loaders for a 64-wide LDS slab (ld = 64).
__device__ __forceinline__ v16bf lds_a_frag(const __bf16* base, int m0, int k0, int lane) {
    const __bf16* p = base + (m0 + (lane & 15)) * 64 + k0 + ((lane >> 4) << 4);
    FragBF f;
    f.q[0] = *(const uint4*)p;
    f.q[1] = *(const uint4*)(p + 8);
    return f.v;
}
__device__ __forceinline__ v16bf lds_b_frag(const __bf16* base, int n0, int k0, int lane) {
    const __bf16* p = base + (n0 + (lane & 15)) * 64 + k0 + ((lane >> 4) << 4);
    FragBF f;
    f.q[0] = *(const uint4*)p;
    f.q[1] = *(const uint4*)(p + 8);
    return f.v;
}

__device__ __forceinline__ int rel_bucket(int j, int i) {
    int d = j - i;
    d = d < -MAX_REL ? -MAX_REL : d;
    d = d >  MAX_REL ?  MAX_REL : d;
    return d + MAX_REL;
}

// ---------------------------------------------------------------------------
// f32 -> bf16 convert (vectorized by 4)
// ---------------------------------------------------------------------------
__global__ void cvt_bf16(const float4* __restrict__ X, __bf16* __restrict__ O, int n4) {
    int i = blockIdx.x * 256 + threadIdx.x;
    if (i < n4) {
        float4 v = X[i];
        union { __bf16 h[4]; uint2 u; } t;
        t.h[0] = (__bf16)v.x; t.h[1] = (__bf16)v.y;
        t.h[2] = (__bf16)v.z; t.h[3] = (__bf16)v.w;
        ((uint2*)O)[i] = t.u;
    }
}

// ---------------------------------------------------------------------------
// Row LayerNorm (D_MODEL = 1024), bf16 output. One block per row.
// ---------------------------------------------------------------------------
__global__ void __launch_bounds__(256)
layernorm_bf16(const float* __restrict__ X, const float* __restrict__ G,
               const float* __restrict__ Bt, __bf16* __restrict__ O) {
    int row = blockIdx.x;
    const float* xr = X + (size_t)row * D_MODEL;
    float s = 0.f, s2 = 0.f;
    for (int i = threadIdx.x; i < D_MODEL; i += 256) {
        float v = xr[i]; s += v; s2 += v * v;
    }
    for (int m = 16; m; m >>= 1) { s += __shfl_xor(s, m); s2 += __shfl_xor(s2, m); }
    __shared__ float red[16];
    __shared__ float stat[2];
    int w = threadIdx.x >> 5;
    if ((threadIdx.x & 31) == 0) { red[w] = s; red[8 + w] = s2; }
    __syncthreads();
    if (threadIdx.x == 0) {
        float a = 0.f, b = 0.f;
        for (int i = 0; i < 8; ++i) { a += red[i]; b += red[8 + i]; }
        float mu  = a / (float)D_MODEL;
        float var = b / (float)D_MODEL - mu * mu;
        stat[0] = mu; stat[1] = rsqrtf(var + 1e-5f);
    }
    __syncthreads();
    float mu = stat[0], rstd = stat[1];
    for (int i = threadIdx.x; i < D_MODEL; i += 256)
        O[(size_t)row * D_MODEL + i] = (__bf16)((xr[i] - mu) * rstd * G[i] + Bt[i]);
}

// ---------------------------------------------------------------------------
// WMMA GEMM: C = A(MxK) @ W(NxK)^T.  Block 256 thr = 8 waves, tile 128x64.
// A(128x64-slab) and W(64x64-slab) panels are async-staged into LDS, double
// buffered; each wave computes 64x16 from ds_load_b128 fragments.
// EPI=0: bf16 out.  EPI=1: f32 out = acc + bias[n] + res[m][n].
// ---------------------------------------------------------------------------
template <int EPI>
__global__ void __launch_bounds__(256)
gemm_bf16(const __bf16* __restrict__ A, const __bf16* __restrict__ W,
          const float* __restrict__ bias, const float* __restrict__ res,
          void* __restrict__ outv, int M, int N, int K) {
    __shared__ __bf16 sA[2][128 * 64];   // 16 KB per buffer
    __shared__ __bf16 sW[2][64 * 64];    //  8 KB per buffer

    const int tid  = threadIdx.x;
    const int lane = tid & 31;
    const int w    = tid >> 5;
    const int mblk = blockIdx.y * 128;
    const int nblk = blockIdx.x * 64;
    const int mloc = (w >> 2) * 64;
    const int nloc = (w & 3) * 16;

    // Stage one 64-wide K slab: 6 async b128 per thread (4 for A, 2 for W).
    auto issue_slab = [&](int k0, int buf) {
#pragma unroll
        for (int i = 0; i < 4; ++i) {
            int c = tid + 256 * i;
            int row = c >> 3, col = (c & 7) * 8;
            async_ld16(A + (size_t)(mblk + row) * K + k0 + col,
                       &sA[buf][row * 64 + col]);
        }
#pragma unroll
        for (int i = 0; i < 2; ++i) {
            int c = tid + 256 * i;
            int row = c >> 3, col = (c & 7) * 8;
            async_ld16(W + (size_t)(nblk + row) * K + k0 + col,
                       &sW[buf][row * 64 + col]);
        }
    };

    v8f acc0 = {}, acc1 = {}, acc2 = {}, acc3 = {};
    const int nslab = K >> 6;
    issue_slab(0, 0);
    for (int s = 0; s < nslab; ++s) {
        const int buf = s & 1;
        if (s + 1 < nslab) { issue_slab((s + 1) << 6, buf ^ 1); wait_async<6>(); }
        else               { wait_async<0>(); }
        __syncthreads();
#pragma unroll
        for (int kk = 0; kk < 64; kk += 32) {
            v16bf b  = lds_b_frag(&sW[buf][0], nloc, kk, lane);
            v16bf a0 = lds_a_frag(&sA[buf][0], mloc +  0, kk, lane);
            v16bf a1 = lds_a_frag(&sA[buf][0], mloc + 16, kk, lane);
            v16bf a2 = lds_a_frag(&sA[buf][0], mloc + 32, kk, lane);
            v16bf a3 = lds_a_frag(&sA[buf][0], mloc + 48, kk, lane);
            acc0 = wmma_bf16(a0, b, acc0);
            acc1 = wmma_bf16(a1, b, acc1);
            acc2 = wmma_bf16(a2, b, acc2);
            acc3 = wmma_bf16(a3, b, acc3);
        }
        __syncthreads();   // all waves done with buf before it is refilled
    }

    int n  = nblk + nloc + (lane & 15);
    int mr = mblk + mloc + ((lane >> 4) << 3);
    v8f accs[4] = {acc0, acc1, acc2, acc3};
#pragma unroll
    for (int t = 0; t < 4; ++t) {
#pragma unroll
        for (int r = 0; r < 8; ++r) {
            size_t idx = (size_t)(mr + t * 16 + r) * N + n;
            float vv = accs[t][r];
            if (EPI == 0) {
                ((__bf16*)outv)[idx] = (__bf16)vv;
            } else {
                ((float*)outv)[idx] = vv + bias[n] + res[idx];
            }
        }
    }
}

// ---------------------------------------------------------------------------
// FFN1 with fused GLU: gate = Y@W1[0:FF]^T + b1[0:FF], lin = Y@W1[FF:2FF]^T +
// b1[FF:2FF]; out = gelu_exact(gate) * lin (bf16). A panel async-staged in
// LDS (shared by all 8 waves); dual W panels stay direct-from-global.
// ---------------------------------------------------------------------------
__global__ void __launch_bounds__(256)
ffn1_glu(const __bf16* __restrict__ Y, const __bf16* __restrict__ W1,
         const float* __restrict__ b1, __bf16* __restrict__ FF) {
    __shared__ __bf16 sA[2][128 * 64];

    const int tid  = threadIdx.x;
    const int lane = tid & 31;
    const int w    = tid >> 5;
    const int mblk = blockIdx.y * 128;
    const int n0   = blockIdx.x * 64 + (w & 3) * 16;
    const int mloc = (w >> 2) * 64;

    auto issue_slab = [&](int k0, int buf) {
#pragma unroll
        for (int i = 0; i < 4; ++i) {
            int c = tid + 256 * i;
            int row = c >> 3, col = (c & 7) * 8;
            async_ld16(Y + (size_t)(mblk + row) * D_MODEL + k0 + col,
                       &sA[buf][row * 64 + col]);
        }
    };

    v8f g0 = {}, g1 = {}, g2 = {}, g3 = {};
    v8f l0 = {}, l1 = {}, l2 = {}, l3 = {};
    issue_slab(0, 0);
    for (int s = 0; s < D_MODEL / 64; ++s) {
        const int buf = s & 1;
        if (s + 1 < D_MODEL / 64) { issue_slab((s + 1) << 6, buf ^ 1); wait_async<4>(); }
        else                      { wait_async<0>(); }
        __syncthreads();
#pragma unroll
        for (int kk = 0; kk < 64; kk += 32) {
            int k0 = (s << 6) + kk;
            __builtin_prefetch(W1 + (size_t)(n0 + (lane & 15)) * D_MODEL + k0 + 64, 0, 1);
            v16bf bg = load_b_frag(W1, D_MODEL, n0,        k0, lane);
            v16bf bl = load_b_frag(W1, D_MODEL, n0 + D_FF, k0, lane);
            v16bf a0 = lds_a_frag(&sA[buf][0], mloc +  0, kk, lane);
            v16bf a1 = lds_a_frag(&sA[buf][0], mloc + 16, kk, lane);
            v16bf a2 = lds_a_frag(&sA[buf][0], mloc + 32, kk, lane);
            v16bf a3 = lds_a_frag(&sA[buf][0], mloc + 48, kk, lane);
            g0 = wmma_bf16(a0, bg, g0); l0 = wmma_bf16(a0, bl, l0);
            g1 = wmma_bf16(a1, bg, g1); l1 = wmma_bf16(a1, bl, l1);
            g2 = wmma_bf16(a2, bg, g2); l2 = wmma_bf16(a2, bl, l2);
            g3 = wmma_bf16(a3, bg, g3); l3 = wmma_bf16(a3, bl, l3);
        }
        __syncthreads();
    }

    int n  = n0 + (lane & 15);
    int mr = mblk + mloc + ((lane >> 4) << 3);
    float bg = b1[n], blin = b1[n + D_FF];
    v8f gg[4] = {g0, g1, g2, g3};
    v8f ll[4] = {l0, l1, l2, l3};
#pragma unroll
    for (int t = 0; t < 4; ++t) {
#pragma unroll
        for (int r = 0; r < 8; ++r) {
            float gate = gg[t][r] + bg;
            float lin  = ll[t][r] + blin;
            float ge   = 0.5f * gate * (1.0f + erff(gate * 0.70710678118654752f));
            FF[(size_t)(mr + t * 16 + r) * D_FF + n] = (__bf16)(ge * lin);
        }
    }
}

// ---------------------------------------------------------------------------
// Two-pass flash attention with relative-position K bias and V output term.
//   qrel[i][r] = q_i . rel_k[r]       (bias gathered per score element)
//   wsum[i][r] = sum_j p_ij over rel bucket r (ds_add_f32), epilogue adds
//   wsum @ rel_v to the WMMA-accumulated P @ V context.
// Block = 128 thr (4 waves); wave owns 16 q-rows; j-tiles of 32.
// ---------------------------------------------------------------------------
__global__ void __launch_bounds__(128)
flash_attn(const __bf16* __restrict__ Q, const __bf16* __restrict__ Kb,
           const __bf16* __restrict__ Vb, const float* __restrict__ relk,
           const float* __restrict__ relv, __bf16* __restrict__ ctx) {
    const int bh = blockIdx.y;
    const int b  = bh >> 4;
    const int h  = bh & 15;
    const int q0 = blockIdx.x * 64;
    const int tid  = threadIdx.x;
    const int lane = tid & 31;
    const int w    = tid >> 5;

    __shared__ float  qrel[64][NREL + 1];   // also reused to stage rel_v
    __shared__ float  wsum[64][NREL + 1];
    __shared__ __bf16 vt[D_K * 32];         // V^T tile: vt[d*32 + jj]
    __shared__ __bf16 pbuf[4][16 * 32];     // per-wave P staging (C->A relayout)

    // qrel + wsum init
    for (int idx = tid; idx < 64 * NREL; idx += 128) {
        int i = idx / NREL, r = idx - i * NREL;
        const __bf16* qp = Q + (size_t)(b * SEQ + q0 + i) * D_MODEL + h * D_K;
        float acc = 0.f;
#pragma unroll 16
        for (int d = 0; d < D_K; ++d) acc += (float)qp[d] * relk[r * D_K + d];
        qrel[i][r] = acc;
        wsum[i][r] = 0.f;
    }
    __syncthreads();

    // Q fragments for this wave's 16 rows (K=64 -> two 16x32 A fragments)
    v16bf qf0, qf1;
    {
        const __bf16* qp = Q + (size_t)(b * SEQ + q0 + w * 16 + (lane & 15)) * D_MODEL
                             + h * D_K + ((lane >> 4) << 4);
        FragBF f;
        f.q[0] = *(const uint4*)qp;        f.q[1] = *(const uint4*)(qp + 8); qf0 = f.v;
        f.q[0] = *(const uint4*)(qp + 32); f.q[1] = *(const uint4*)(qp + 40); qf1 = f.v;
    }

    const int ilBase = w * 16 + ((lane >> 4) << 3);   // first of this lane's 8 rows
    float mrow[8], lrow[8];
#pragma unroll
    for (int r = 0; r < 8; ++r) { mrow[r] = -1e30f; lrow[r] = 0.f; }

    // ---------------- pass 1: softmax stats ----------------
    for (int j0 = 0; j0 < SEQ; j0 += 32) {
#pragma unroll
        for (int nt = 0; nt < 2; ++nt) {
            int j = j0 + nt * 16 + (lane & 15);
            const __bf16* kp = Kb + (size_t)(b * SEQ + j) * D_MODEL + h * D_K
                                  + ((lane >> 4) << 4);
            FragBF f;
            f.q[0] = *(const uint4*)kp;        f.q[1] = *(const uint4*)(kp + 8);
            v8f s = {};
            s = wmma_bf16(qf0, f.v, s);
            f.q[0] = *(const uint4*)(kp + 32); f.q[1] = *(const uint4*)(kp + 40);
            s = wmma_bf16(qf1, f.v, s);
#pragma unroll
            for (int r = 0; r < 8; ++r) {
                int il = ilBase + r;
                float sv = s[r] * 0.125f + qrel[il][rel_bucket(j, q0 + il)];
                float mj = sv;
                for (int msk = 8; msk; msk >>= 1) mj = fmaxf(mj, __shfl_xor(mj, msk));
                float mnew = fmaxf(mrow[r], mj);
                float p  = __expf(sv - mnew);
                float ps = p;
                for (int msk = 8; msk; msk >>= 1) ps += __shfl_xor(ps, msk);
                lrow[r] = lrow[r] * __expf(mrow[r] - mnew) + ps;
                mrow[r] = mnew;
            }
        }
    }
    float linv[8];
#pragma unroll
    for (int r = 0; r < 8; ++r) linv[r] = 1.0f / lrow[r];

    // ---------------- pass 2: P@V + rel bucket sums ----------------
    v8f c0 = {}, c1 = {}, c2 = {}, c3 = {};
    for (int j0 = 0; j0 < SEQ; j0 += 32) {
        __syncthreads();   // previous vt consumers done
        {   // stage V^T tile: 32 j-rows x 64 d-cols
            int jj = tid >> 2, d0 = (tid & 3) * 16;
            const __bf16* vp = Vb + (size_t)(b * SEQ + j0 + jj) * D_MODEL + h * D_K + d0;
            FragBF f;
            f.q[0] = *(const uint4*)vp; f.q[1] = *(const uint4*)(vp + 8);
#pragma unroll
            for (int e = 0; e < 16; ++e) vt[(d0 + e) * 32 + jj] = f.h[e];
        }
        __syncthreads();

#pragma unroll
        for (int nt = 0; nt < 2; ++nt) {
            int j = j0 + nt * 16 + (lane & 15);
            const __bf16* kp = Kb + (size_t)(b * SEQ + j) * D_MODEL + h * D_K
                                  + ((lane >> 4) << 4);
            FragBF f;
            f.q[0] = *(const uint4*)kp;        f.q[1] = *(const uint4*)(kp + 8);
            v8f s = {};
            s = wmma_bf16(qf0, f.v, s);
            f.q[0] = *(const uint4*)(kp + 32); f.q[1] = *(const uint4*)(kp + 40);
            s = wmma_bf16(qf1, f.v, s);
#pragma unroll
            for (int r = 0; r < 8; ++r) {
                int il  = ilBase + r;
                int rb  = rel_bucket(j, q0 + il);
                float sv = s[r] * 0.125f + qrel[il][rb];
                float p  = __expf(sv - mrow[r]) * linv[r];
                atomicAdd(&wsum[il][rb], p);                 // ds_add_f32
                pbuf[w][(((lane >> 4) << 3) + r) * 32 + nt * 16 + (lane & 15)] = (__bf16)p;
            }
        }
        // Re-layout P (C-layout -> A-fragment) through LDS; DS is in-order per wave.
        FragBF pf;
        {
            const __bf16* pp = &pbuf[w][(lane & 15) * 32 + ((lane >> 4) << 4)];
            pf.q[0] = *(const uint4*)pp;
            pf.q[1] = *(const uint4*)(pp + 8);
        }
        v8f* cc[4] = {&c0, &c1, &c2, &c3};
#pragma unroll
        for (int dt = 0; dt < 4; ++dt) {
            const __bf16* vtp = &vt[(dt * 16 + (lane & 15)) * 32 + ((lane >> 4) << 4)];
            FragBF vf;
            vf.q[0] = *(const uint4*)vtp;
            vf.q[1] = *(const uint4*)(vtp + 8);
            *cc[dt] = wmma_bf16(pf.v, vf.v, *cc[dt]);
        }
    }

    // ---------------- epilogue: ctx += wsum @ rel_v ----------------
    __syncthreads();
    float* rvs = &qrel[0][0];   // reuse qrel LDS (4224 floats >= 65*64)
    for (int idx = tid; idx < NREL * D_K; idx += 128) rvs[idx] = relv[idx];
    __syncthreads();

    v8f cacc[4] = {c0, c1, c2, c3};
#pragma unroll
    for (int dt = 0; dt < 4; ++dt) {
#pragma unroll
        for (int r = 0; r < 8; ++r) {
            int il = ilBase + r;
            int d  = dt * 16 + (lane & 15);
            float acc = cacc[dt][r];
            for (int rr = 0; rr < NREL; ++rr)
                acc += wsum[il][rr] * rvs[rr * D_K + d];
            ctx[(size_t)(b * SEQ + q0 + il) * D_MODEL + h * D_K + d] = (__bf16)acc;
        }
    }
}

// ---------------------------------------------------------------------------
// Host orchestration
// ---------------------------------------------------------------------------
extern "C" void kernel_launch(void* const* d_in, const int* in_sizes, int n_in,
                              void* d_out, int out_size, void* d_ws, size_t ws_size,
                              hipStream_t stream) {
    (void)in_sizes; (void)n_in; (void)out_size; (void)ws_size;
    const float* src  = (const float*)d_in[0];
    const float* w_q  = (const float*)d_in[1];
    const float* w_k  = (const float*)d_in[2];
    const float* w_v  = (const float*)d_in[3];
    const float* w_o  = (const float*)d_in[4];
    const float* b_o  = (const float*)d_in[5];
    const float* relk = (const float*)d_in[6];
    const float* relv = (const float*)d_in[7];
    const float* w1   = (const float*)d_in[8];
    const float* b1   = (const float*)d_in[9];
    const float* w2   = (const float*)d_in[10];
    const float* b2   = (const float*)d_in[11];
    const float* g1   = (const float*)d_in[12];
    const float* be1  = (const float*)d_in[13];
    const float* g2   = (const float*)d_in[14];
    const float* be2  = (const float*)d_in[15];
    float* out = (float*)d_out;

    char* ws = (char*)d_ws;
    size_t off = 0;
    auto take = [&](size_t bytes) {
        void* p = ws + off;
        off += (bytes + 255) & ~(size_t)255;
        return p;
    };
    __bf16* xb   = (__bf16*)take((size_t)M_ROWS * D_MODEL * 2);
    __bf16* wqb  = (__bf16*)take((size_t)D_MODEL * D_MODEL * 2);
    __bf16* wkb  = (__bf16*)take((size_t)D_MODEL * D_MODEL * 2);
    __bf16* wvb  = (__bf16*)take((size_t)D_MODEL * D_MODEL * 2);
    __bf16* wob  = (__bf16*)take((size_t)D_MODEL * D_MODEL * 2);
    __bf16* w1b  = (__bf16*)take((size_t)2 * D_FF * D_MODEL * 2);
    __bf16* w2b  = (__bf16*)take((size_t)D_MODEL * D_FF * 2);
    __bf16* qb   = (__bf16*)take((size_t)M_ROWS * D_MODEL * 2);
    __bf16* kb   = (__bf16*)take((size_t)M_ROWS * D_MODEL * 2);
    __bf16* vb   = (__bf16*)take((size_t)M_ROWS * D_MODEL * 2);
    __bf16* ctxb = (__bf16*)take((size_t)M_ROWS * D_MODEL * 2);
    float*  src2 = (float*) take((size_t)M_ROWS * D_MODEL * 4);
    __bf16* yb   = (__bf16*)take((size_t)M_ROWS * D_MODEL * 2);
    __bf16* ffb  = (__bf16*)take((size_t)M_ROWS * D_FF * 2);

    auto cvt = [&](const float* s, __bf16* d, int n) {
        cvt_bf16<<<dim3(n / 1024), dim3(256), 0, stream>>>((const float4*)s, d, n / 4);
    };
    cvt(w_q, wqb, D_MODEL * D_MODEL);
    cvt(w_k, wkb, D_MODEL * D_MODEL);
    cvt(w_v, wvb, D_MODEL * D_MODEL);
    cvt(w_o, wob, D_MODEL * D_MODEL);
    cvt(w1,  w1b, 2 * D_FF * D_MODEL);
    cvt(w2,  w2b, D_MODEL * D_FF);

    // LN1 -> x (bf16)
    layernorm_bf16<<<dim3(M_ROWS), dim3(256), 0, stream>>>(src, g1, be1, xb);

    // Q, K, V projections
    dim3 gproj(D_MODEL / 64, M_ROWS / 128);
    gemm_bf16<0><<<gproj, dim3(256), 0, stream>>>(xb, wqb, nullptr, nullptr, qb,
                                                  M_ROWS, D_MODEL, D_MODEL);
    gemm_bf16<0><<<gproj, dim3(256), 0, stream>>>(xb, wkb, nullptr, nullptr, kb,
                                                  M_ROWS, D_MODEL, D_MODEL);
    gemm_bf16<0><<<gproj, dim3(256), 0, stream>>>(xb, wvb, nullptr, nullptr, vb,
                                                  M_ROWS, D_MODEL, D_MODEL);

    // Attention (rel-pos flash, two-pass)
    flash_attn<<<dim3(SEQ / 64, BATCH * N_HEADS), dim3(128), 0, stream>>>(
        qb, kb, vb, relk, relv, ctxb);

    // Output projection + b_o + residual(src) -> src2 (f32)
    gemm_bf16<1><<<gproj, dim3(256), 0, stream>>>(ctxb, wob, b_o, src, src2,
                                                  M_ROWS, D_MODEL, D_MODEL);

    // LN2 -> y (bf16)
    layernorm_bf16<<<dim3(M_ROWS), dim3(256), 0, stream>>>(src2, g2, be2, yb);

    // FFN1 + fused GLU -> ff (bf16)
    ffn1_glu<<<dim3(D_FF / 64, M_ROWS / 128), dim3(256), 0, stream>>>(yb, w1b, b1, ffb);

    // FFN2 + b2 + residual(src2) -> out (f32)
    gemm_bf16<1><<<gproj, dim3(256), 0, stream>>>(ffb, w2b, b2, src2, out,
                                                  M_ROWS, D_MODEL, D_FF);
}
